// FactorGraphConvolution_33535104647627
// MI455X (gfx1250) — matrix operand order repaired and checked by reading, not compiled, not run on hardware
//
#include <hip/hip_runtime.h>

// ---------------------------------------------------------------------------
// FactorGraphConvolution for MI455X (gfx1250, wave32, WMMA bf16 path)
//   N = 8192, IN_F = 256, OUT_F = 256
// Dominant work: 3x [8192 x 8192] @ [8192 x 256] matmuls (~103 GFLOP) over
// 512 MB of adjacency data (HBM floor ~22us @ 23.3 TB/s).
//   - v_wmma_f32_16x16x32_bf16 for all matmuls (masks are exact in bf16)
//   - global_load_async_to_lds_b128 + ASYNCcnt double buffering for the
//     adjacency stream (one barrier per 64-wide K step)
//   - 32-row strips / 64 columns per wave so each A-fragment build (the
//     VALU-heavy mask construction) feeds 12 WMMAs
// ---------------------------------------------------------------------------

typedef __attribute__((ext_vector_type(16))) __bf16 v16bf;
typedef __attribute__((ext_vector_type(8)))  __bf16 v8bf;
typedef __attribute__((ext_vector_type(8)))  float  v8f;

#define GN   8192
#define GIN  256
#define GOUT 256

// ---------------------------------------------------------------------------
// Kernel 0: one-shot conversions to bf16 (transposed so WMMA B-fragments are
// contiguous 32B per-lane loads). All small; L2 absorbs the scattered writes.
// ---------------------------------------------------------------------------
__global__ __launch_bounds__(256) void fgc_convert_kernel(
    const float* __restrict__ feats,
    const float* __restrict__ node_w,
    const float* __restrict__ edge_w,
    __bf16* __restrict__ fT,
    __bf16* __restrict__ nwT,
    __bf16* __restrict__ ewT)
{
  const int idx = blockIdx.x * 256 + threadIdx.x;
  if (idx < GN * GIN) {
    const int k = idx >> 8;        // row of feats
    const int n = idx & 255;       // col of feats
    fT[(size_t)n * GN + k] = (__bf16)feats[idx];
  }
  if (idx < 2 * GIN * GOUT) {
    const int k = idx >> 8;        // row of node_weight (0..511)
    const int j = idx & 255;       // col (0..255)
    nwT[(size_t)j * (2 * GIN) + k] = (__bf16)node_w[idx];
  }
  if (idx < GIN * GOUT) {
    const int k = idx >> 8;
    const int j = idx & 255;
    ewT[(size_t)j * GIN + k] = (__bf16)edge_w[idx];
  }
}

// Async copy of 64 contiguous bytes (global -> LDS), ASYNCcnt-tracked.
// INST_OFFSET applies to both the global and LDS addresses (ISA 10.7/15.18),
// so one address pair covers all four 16B slices.
__device__ __forceinline__ void async_copy64(const float* gsrc, float* ldst) {
  const unsigned lds_off = (unsigned)(unsigned long long)(uintptr_t)ldst;
  asm volatile("global_load_async_to_lds_b128 %0, %1, off"
               :: "v"(lds_off), "v"(gsrc) : "memory");
  asm volatile("global_load_async_to_lds_b128 %0, %1, off offset:16"
               :: "v"(lds_off), "v"(gsrc) : "memory");
  asm volatile("global_load_async_to_lds_b128 %0, %1, off offset:32"
               :: "v"(lds_off), "v"(gsrc) : "memory");
  asm volatile("global_load_async_to_lds_b128 %0, %1, off offset:48"
               :: "v"(lds_off), "v"(gsrc) : "memory");
}

// ---------------------------------------------------------------------------
// Kernel 1: the three N x N x 256 matmuls, fused.
// Grid: 256 blocks (one 32-row strip each). Block: 256 threads = 8 waves.
// Waves 0-3: rows [row0, row0+16), waves 4-7: rows [row0+16, row0+32).
// Each wave owns 64 feature columns = 4 WMMA col tiles, so one A-fragment
// build feeds 12 WMMAs. K-step of 64 with double-buffered async LDS staging.
// ---------------------------------------------------------------------------
__global__ __launch_bounds__(256) void fgc_spmm_kernel(
    const float* __restrict__ node_adj,   // [GN][GN]
    const float* __restrict__ edge_adj,   // [GN][GN]
    const __bf16* __restrict__ fT,        // [GIN][GN]
    __bf16* __restrict__ node_support,    // [GN][2*GIN]  (pos | neg)
    __bf16* __restrict__ edge_support)    // [GN][GIN]
{
  __shared__ float lds_node[2][32][64];   // 16 KB
  __shared__ float lds_edge[2][32][64];   // 16 KB

  const int row0 = blockIdx.x * 32;
  const int tid  = threadIdx.x;
  const int lane = tid & 31;
  const int wave = tid >> 5;

  const int rh   = wave >> 2;          // row half: 0 or 1
  const int nb   = (wave & 3) * 64;    // this wave's first feature column
  const int rowb = row0 + rh * 16;     // this wave's 16-row tile base

  const int m   = lane & 15;           // A row / B column within tile
  const int hw  = lane >> 4;           // half-wave id (0/1)
  const int ka0 = hw * 16;             // A layout: lane<16 -> K 0..15
  const int ka1 = ka0 + 8;             //           lane>=16 -> K 16..31
  const int kb  = hw * 16;             // B layout: lanes 0-15 K 0..15, 16-31 K 16..31
  const int am  = rh * 16 + m;         // this lane's row within the 32-row tile

  // Async staging roles: waves 0-3 stream node_adj, waves 4-7 stream edge_adj.
  // 128 threads per matrix, each moving one 64B slice of the 32x64 f32 tile.
  const int  tt      = tid & 127;
  const int  amm     = tt >> 2;        // tile row 0..31
  const int  akk     = (tt & 3) * 16;  // tile col 0,16,32,48 (16 floats = 64B)
  const bool is_node = (tid < 128);
  const float* gbase =
      (is_node ? node_adj : edge_adj) + (size_t)(row0 + amm) * GN + akk;

  v8f acc_pos[4]  = {};
  v8f acc_neg[4]  = {};
  v8f acc_edge[4] = {};

  const __bf16 one  = (__bf16)1.0f;
  const __bf16 zero = (__bf16)0.0f;

  // prologue: prefetch buffer 0
  {
    float* lp = is_node ? &lds_node[0][amm][akk] : &lds_edge[0][amm][akk];
    async_copy64(gbase, lp);
  }

  for (int k0 = 0; k0 < GN; k0 += 64) {
    const int buf = (k0 >> 6) & 1;

    asm volatile("s_wait_asynccnt 0" ::: "memory");  // own async writes landed
    __syncthreads();  // all waves: buf complete, and everyone past buf^1 reads

    // prefetch next tile into the buffer everyone just finished reading
    if (k0 + 64 < GN) {
      float* lp = is_node ? &lds_node[buf ^ 1][amm][akk]
                          : &lds_edge[buf ^ 1][amm][akk];
      async_copy64(gbase + k0 + 64, lp);
    }

#pragma unroll
    for (int kh = 0; kh < 2; ++kh) {
      const int ks = kh * 32;

      // ---- build three bf16 A fragments (16x32) from LDS ----
      v16bf a_pos, a_neg, a_edge;
#pragma unroll
      for (int h = 0; h < 8; ++h) {
        const float vn0 = lds_node[buf][am][ks + ka0 + h];
        const float vn1 = lds_node[buf][am][ks + ka1 + h];
        a_pos[h]     = (vn0 > 0.0f) ? one : zero;
        a_pos[8 + h] = (vn1 > 0.0f) ? one : zero;
        a_neg[h]     = (vn0 < 0.0f) ? one : zero;
        a_neg[8 + h] = (vn1 < 0.0f) ? one : zero;
        a_edge[h]     = (__bf16)lds_edge[buf][am][ks + ka0 + h];
        a_edge[8 + h] = (__bf16)lds_edge[buf][am][ks + ka1 + h];
      }

      // ---- B fragments from L2-resident transposed feats + 12 WMMAs ----
#pragma unroll
      for (int ct = 0; ct < 4; ++ct) {
        const int n = nb + ct * 16 + m;
        const v16bf b = *(const v16bf*)(fT + (size_t)n * GN + k0 + ks + kb);
        acc_pos[ct] = __builtin_amdgcn_wmma_f32_16x16x32_bf16(
            false, a_pos, false, b, (short)0, acc_pos[ct], false, false);
        acc_neg[ct] = __builtin_amdgcn_wmma_f32_16x16x32_bf16(
            false, a_neg, false, b, (short)0, acc_neg[ct], false, false);
        acc_edge[ct] = __builtin_amdgcn_wmma_f32_16x16x32_bf16(
            false, a_edge, false, b, (short)0, acc_edge[ct], false, false);
      }
    }
  }

  // ---- epilogue: C tile layout -> row-major bf16 support matrices ----
#pragma unroll
  for (int ct = 0; ct < 4; ++ct) {
    const int col = nb + ct * 16 + m;
#pragma unroll
    for (int r = 0; r < 8; ++r) {
      const int row = rowb + hw * 8 + r;
      node_support[(size_t)row * (2 * GIN) + col]       = (__bf16)acc_pos[ct][r];
      node_support[(size_t)row * (2 * GIN) + GIN + col] = (__bf16)acc_neg[ct][r];
      edge_support[(size_t)row * GIN + col]             = (__bf16)acc_edge[ct][r];
    }
  }
}

// ---------------------------------------------------------------------------
// Kernel 2: out = node_support @ W_n + edge_support @ W_e
//               + node_bias * diag(edge_adj) + edge_bias * diag(node_adj)
// Small (~3 GFLOP); same tiling, WMMA over K=512 then K=256.
// ---------------------------------------------------------------------------
__global__ __launch_bounds__(256) void fgc_out_kernel(
    const __bf16* __restrict__ node_support, // [GN][2*GIN]
    const __bf16* __restrict__ edge_support, // [GN][GIN]
    const __bf16* __restrict__ nwT,          // [GOUT][2*GIN]
    const __bf16* __restrict__ ewT,          // [GOUT][GIN]
    const float* __restrict__ node_bias,     // [GOUT]
    const float* __restrict__ edge_bias,     // [GOUT]
    const float* __restrict__ node_adj,      // [GN][GN] (diag only)
    const float* __restrict__ edge_adj,      // [GN][GN] (diag only)
    float* __restrict__ out)                 // [GN][GOUT]
{
  const int row0 = blockIdx.x * 16;
  const int lane = threadIdx.x & 31;
  const int wave = threadIdx.x >> 5;
  const int nb   = wave * 32;

  const int m   = lane & 15;
  const int hw  = lane >> 4;
  const int ka0 = hw * 16;
  const int ka1 = ka0 + 8;
  const int kb  = hw * 16;

  v8f acc[2] = {};

  // node part: K = 512
  for (int k0 = 0; k0 < 2 * GIN; k0 += 32) {
    const __bf16* ap = node_support + (size_t)(row0 + m) * (2 * GIN) + k0;
    const v8bf alo = *(const v8bf*)(ap + ka0);
    const v8bf ahi = *(const v8bf*)(ap + ka1);
    v16bf a;
#pragma unroll
    for (int h = 0; h < 8; ++h) { a[h] = alo[h]; a[8 + h] = ahi[h]; }
#pragma unroll
    for (int ct = 0; ct < 2; ++ct) {
      const int j = nb + ct * 16 + m;
      const v16bf b = *(const v16bf*)(nwT + (size_t)j * (2 * GIN) + k0 + kb);
      acc[ct] = __builtin_amdgcn_wmma_f32_16x16x32_bf16(
          false, a, false, b, (short)0, acc[ct], false, false);
    }
  }

  // edge part: K = 256
  for (int k0 = 0; k0 < GIN; k0 += 32) {
    const __bf16* ap = edge_support + (size_t)(row0 + m) * GIN + k0;
    const v8bf alo = *(const v8bf*)(ap + ka0);
    const v8bf ahi = *(const v8bf*)(ap + ka1);
    v16bf a;
#pragma unroll
    for (int h = 0; h < 8; ++h) { a[h] = alo[h]; a[8 + h] = ahi[h]; }
#pragma unroll
    for (int ct = 0; ct < 2; ++ct) {
      const int j = nb + ct * 16 + m;
      const v16bf b = *(const v16bf*)(ewT + (size_t)j * GIN + k0 + kb);
      acc[ct] = __builtin_amdgcn_wmma_f32_16x16x32_bf16(
          false, a, false, b, (short)0, acc[ct], false, false);
    }
  }

  // epilogue: bias * diagonal terms, f32 output
#pragma unroll
  for (int r = 0; r < 8; ++r) {
    const int row = row0 + hw * 8 + r;
    const float dE = edge_adj[(size_t)row * (GN + 1)];
    const float dN = node_adj[(size_t)row * (GN + 1)];
#pragma unroll
    for (int ct = 0; ct < 2; ++ct) {
      const int col = nb + ct * 16 + m;
      out[(size_t)row * GOUT + col] =
          acc[ct][r] + node_bias[col] * dE + edge_bias[col] * dN;
    }
  }
}

// ---------------------------------------------------------------------------
extern "C" void kernel_launch(void* const* d_in, const int* in_sizes, int n_in,
                              void* d_out, int out_size, void* d_ws, size_t ws_size,
                              hipStream_t stream) {
  const float* feats    = (const float*)d_in[0];
  const float* node_adj = (const float*)d_in[1];
  const float* edge_adj = (const float*)d_in[2];
  const float* node_w   = (const float*)d_in[3];
  const float* node_b   = (const float*)d_in[4];
  const float* edge_w   = (const float*)d_in[5];
  const float* edge_b   = (const float*)d_in[6];
  float* out = (float*)d_out;

  // workspace partition (bytes), total ~16.4 MB
  char* ws = (char*)d_ws;
  __bf16* fT   = (__bf16*)ws;                              // 256*8192*2  = 4 MB
  ws += (size_t)GIN * GN * sizeof(__bf16);
  __bf16* nwT  = (__bf16*)ws;                              // 256*512*2   = 256 KB
  ws += (size_t)GOUT * 2 * GIN * sizeof(__bf16);
  __bf16* ewT  = (__bf16*)ws;                              // 256*256*2   = 128 KB
  ws += (size_t)GOUT * GIN * sizeof(__bf16);
  __bf16* node_support = (__bf16*)ws;                      // 8192*512*2  = 8 MB
  ws += (size_t)GN * 2 * GIN * sizeof(__bf16);
  __bf16* edge_support = (__bf16*)ws;                      // 8192*256*2  = 4 MB

  fgc_convert_kernel<<<(GN * GIN + 255) / 256, 256, 0, stream>>>(
      feats, node_w, edge_w, fT, nwT, ewT);

  fgc_spmm_kernel<<<GN / 32, 256, 0, stream>>>(
      node_adj, edge_adj, fT, node_support, edge_support);

  fgc_out_kernel<<<GN / 16, 256, 0, stream>>>(
      node_support, edge_support, nwT, ewT, node_b, edge_b,
      node_adj, edge_adj, out);
}
